// ConvGAU_73254962200660
// MI455X (gfx1250) — matrix operand order repaired and checked by reading, not compiled, not run on hardware
//
#include <hip/hip_runtime.h>
#include <hip/hip_bf16.h>

typedef __attribute__((ext_vector_type(16))) __bf16 v16bf;
typedef __attribute__((ext_vector_type(8)))  float  v8f;
typedef __attribute__((ext_vector_type(4)))  unsigned int u32x4;
typedef __attribute__((ext_vector_type(8)))  int i32x8;
typedef __attribute__((ext_vector_type(4)))  int i32x4;

#define B_SZ  16
#define C_IN  256
#define N_SP  2304      // 48*48 spatial positions
#define HIDD  512
#define QKC   96

#if defined(__has_builtin)
#if __has_builtin(__builtin_amdgcn_tensor_load_to_lds) && \
    __has_builtin(__builtin_amdgcn_s_wait_tensorcnt)
#define USE_TDM 1
#endif
#endif
#ifndef USE_TDM
#define USE_TDM 0
#endif

__device__ __forceinline__ v8f wmma_bf16(v16bf a, v16bf b, v8f c) {
  // D = A(16x32) * B(32x16) + C, f32 accumulate
  return __builtin_amdgcn_wmma_f32_16x16x32_bf16(false, a, false, b, (short)0, c,
                                                 false, false);
}

// A-fragment 16x32 (MxK) from row-major source (stride in elements).
// ISA layout: lanes 0-15: M=lane, elems 0-7 -> K=0..7, elems 8-15 -> K=16..23;
//             lanes 16-31: same M, all K +8.
template <typename T>
__device__ __forceinline__ v16bf load_a(const T* tile, int stride, int lane) {
  const T* row = tile + (size_t)(lane & 15) * stride + ((lane >> 4) & 1) * 8;
  v16bf a;
#pragma unroll
  for (int j = 0; j < 16; ++j) {
    const int kk = (j < 8) ? j : (8 + j);
    a[j] = (__bf16)row[kk];
  }
  return a;
}

// B-fragment 32x16 (KxN) from row-major source: N = lane&15, K in A's table order.
template <typename T>
__device__ __forceinline__ v16bf load_b(const T* tile, int stride, int lane) {
  const T* col = tile + (lane & 15) + (size_t)(((lane >> 4) & 1) * 8) * stride;
  v16bf b;
#pragma unroll
  for (int j = 0; j < 16; ++j) {
    const int kk = (j < 8) ? j : (8 + j);
    b[j] = (__bf16)col[(size_t)kk * stride];
  }
  return b;
}

// B-fragment straight out of an LDS-staged 32x64 row-major tile.
__device__ __forceinline__ v16bf lds_b(const __bf16* sm, int lane, int s) {
  const __bf16* col = sm + s * 16 + (lane & 15) + ((lane >> 4) & 1) * 8 * 64;
  v16bf b;
#pragma unroll
  for (int j = 0; j < 16; ++j) {
    const int kk = (j < 8) ? j : (8 + j);
    b[j] = col[kk * 64];
  }
  return b;
}

// Fallback cooperative stage of a 32x64 bf16 tile into LDS (256 thr x 16B).
__device__ __forceinline__ void stage_tile(__bf16* dst, const __bf16* src, int tid) {
  const int idx = tid * 8;        // 8 bf16 per thread
  const int row = idx >> 6;       // 0..31
  const int col = idx & 63;
  *(uint4*)(dst + idx) = *(const uint4*)(src + (size_t)row * N_SP + col);
}

// For LDS-aperture flat addresses the low 32 bits give the LDS address.
__device__ __forceinline__ unsigned lds_off(const void* p) {
  return (unsigned)(uintptr_t)p;
}

#if USE_TDM
// TDM: DMA a 32-row x 64-col bf16 tile (row stride N_SP elements) from global
// into LDS (contiguous 4 KB). D# built per CDNA5 ISA sec 8.3-8.5:
//  group0: count=1 | lds_addr | global_addr[56:0] | type=2
//  group1: data_size=1(2B), tensor_dim0/1=N_SP, tile_dim0=64, tile_dim1=32,
//          tensor_dim0_stride=N_SP; no padding/iterate/multicast.
//  groups 2/3 (+ trailing group arg of the 6-arg builtin): zero (2-D tensor).
// Wave-level op (EXEC ignored): call from one wave per block only.
__device__ __forceinline__ void tdm_load_tile_32x64(const __bf16* gsrc,
                                                    unsigned lds_byte_addr) {
  const unsigned long long ga = (unsigned long long)(uintptr_t)gsrc;
  u32x4 g0;
  g0[0] = 1u;                                            // count=1, user desc
  g0[1] = lds_byte_addr;                                 // lds_addr
  g0[2] = (unsigned)(ga & 0xffffffffu);                  // global_addr[31:0]
  g0[3] = (unsigned)((ga >> 32) & 0x01ffffffu)           // global_addr[56:32]
          | (2u << 30);                                  // type=2 ("image")
  i32x8 g1;
  g1[0] = (int)(1u << 16);                               // data_size=1 -> 2 bytes
  g1[1] = (int)(((unsigned)N_SP & 0xffffu) << 16);       // tensor_dim0[15:0]
  g1[2] = (int)((((unsigned)N_SP >> 16) & 0xffffu)       // tensor_dim0[31:16]
          | (((unsigned)N_SP & 0xffffu) << 16));         // tensor_dim1[15:0]
  g1[3] = (int)((((unsigned)N_SP >> 16) & 0xffffu)       // tensor_dim1[31:16]
          | (64u << 16));                                // tile_dim0 = 64
  g1[4] = (int)32u;                                      // tile_dim1=32, dim2=0
  g1[5] = (int)(unsigned)N_SP;                           // dim0_stride[31:0]
  g1[6] = 0;                                             // stride high bits
  g1[7] = 0;
  const i32x4 z4 = {0, 0, 0, 0};
  const i32x8 z8 = {0, 0, 0, 0, 0, 0, 0, 0};
  // 6-arg form (clang-23 / therock-10.0 headers).
  __builtin_amdgcn_tensor_load_to_lds(g0, g1, z4, z4, z8, 0);
}
#endif

__device__ __forceinline__ float silu_f(float x) { return x / (1.0f + __expf(-x)); }

// ---------------------------------------------------------------------------
// Kernel 1: fused projections. P = SiLU([w_hidden; w_qk] @ x_b + bias)
// Routes rows: [0,512)->v(bf16), [512,1024)->gate(bf16),
//              [1024,1120)->q stored TRANSPOSED [N][QK], [1120,1216)->k.
// One wave computes 16(M) x 64(N); K = 256 -> 8 WMMA steps x 4 tiles.
// ---------------------------------------------------------------------------
__global__ void __launch_bounds__(32)
proj_kernel(const float* __restrict__ x,
            const float* __restrict__ w_hidden, const float* __restrict__ b_hidden,
            const float* __restrict__ w_qk,     const float* __restrict__ b_qk,
            __bf16* __restrict__ qT, __bf16* __restrict__ kb,
            __bf16* __restrict__ vb, __bf16* __restrict__ gb) {
  const int lane = threadIdx.x;
  const int n0 = blockIdx.x * 64;   // gridDim.x = 36
  const int m0 = blockIdx.y * 16;   // gridDim.y = 76
  const int b  = blockIdx.z;
  const float* xb = x + (size_t)b * C_IN * N_SP;
  const bool is_qk = (m0 >= 2 * HIDD);           // uniform per block
  const float* wsrc = is_qk ? (w_qk + (size_t)(m0 - 2 * HIDD) * C_IN)
                            : (w_hidden + (size_t)m0 * C_IN);
  v8f acc[4] = {};
#pragma unroll
  for (int k0 = 0; k0 < C_IN; k0 += 32) {
    const v16bf a = load_a(wsrc + k0, C_IN, lane);
#pragma unroll
    for (int s = 0; s < 4; ++s)
      acc[s] = wmma_bf16(a, load_b(xb + (size_t)k0 * N_SP + n0 + s * 16, N_SP, lane),
                         acc[s]);
  }
  const int hi = (lane >> 4) & 1;
  const int lo = lane & 15;
#pragma unroll
  for (int s = 0; s < 4; ++s) {
#pragma unroll
    for (int r = 0; r < 8; ++r) {
      const int m = m0 + r + hi * 8;
      const int n = n0 + s * 16 + lo;
      const float bias = is_qk ? b_qk[m - 2 * HIDD] : b_hidden[m];
      const float val = silu_f(acc[s][r] + bias);
      if (m0 < HIDD) {
        vb[((size_t)b * HIDD + m) * N_SP + n] = (__bf16)val;
      } else if (m0 < 2 * HIDD) {
        gb[((size_t)b * HIDD + (m - HIDD)) * N_SP + n] = (__bf16)val;
      } else if (m0 < 2 * HIDD + QKC) {
        qT[((size_t)b * N_SP + n) * QKC + (m - 2 * HIDD)] = (__bf16)val;   // q^T
      } else {
        kb[((size_t)b * QKC + (m - 2 * HIDD - QKC)) * N_SP + n] = (__bf16)val;
      }
    }
  }
}

// --- shared staging helpers used by the three bf16 GEMM kernels -------------
// Prologue stage of tile 0 into smB0, with barrier.
__device__ __forceinline__ void stage_first(__bf16* smB0, const __bf16* src,
                                            int tid, int wv) {
#if USE_TDM
  if (wv == 0) {
    tdm_load_tile_32x64(src, lds_off(smB0));
    __builtin_amdgcn_s_wait_tensorcnt(0);
  }
#else
  stage_tile(smB0, src, tid);
  (void)wv;
#endif
  __syncthreads();
}
// Issue stage of the next tile (overlapped with compute).
__device__ __forceinline__ void stage_next(__bf16* smBn, const __bf16* src,
                                           int tid, int wv) {
#if USE_TDM
  if (wv == 0) tdm_load_tile_32x64(src, lds_off(smBn));
#else
  stage_tile(smBn, src, tid);
  (void)wv;
#endif
}
// Complete the in-flight stage and publish to all waves.
__device__ __forceinline__ void stage_sync(bool pending, int wv) {
#if USE_TDM
  if (wv == 0 && pending) __builtin_amdgcn_s_wait_tensorcnt(0);
#else
  (void)pending; (void)wv;
#endif
  __syncthreads();
}

// ---------------------------------------------------------------------------
// Kernel 2: At[m][n] = relu(q^T k)^2, stored transposed (bf16) to feed the
// V-aggregation GEMM directly. 256-thread block = 8 waves covering 8
// consecutive 32-row n-tiles; the 32x64 k B-tile is staged once per K-step
// (TDM DMA into double-buffered LDS) and shared by all 8 waves. K=96 -> 3 steps.
// ---------------------------------------------------------------------------
__global__ void __launch_bounds__(256)
sim_kernel(const __bf16* __restrict__ qT, const __bf16* __restrict__ kb,
           __bf16* __restrict__ At) {
  const int tid  = threadIdx.x;
  const int lane = tid & 31;
  const int wv   = tid >> 5;                       // 0..7
  const int n0 = blockIdx.x * 256 + wv * 32;       // gridDim.x = 9
  const int m0 = blockIdx.y * 64;                  // gridDim.y = 36
  const int b  = blockIdx.z;
  const __bf16* qTb = qT + (size_t)b * N_SP * QKC;
  const __bf16* kbb = kb + (size_t)b * QKC * N_SP;
  __bf16* Atb = At + (size_t)b * N_SP * N_SP;
  __shared__ __align__(32) __bf16 smB[2][32 * 64];

  v8f acc[2][4] = {};
  stage_first(smB[0], kbb + m0, tid, wv);
#pragma unroll
  for (int c0 = 0; c0 < QKC; c0 += 32) {
    const int cur = (c0 >> 5) & 1;
    const bool more = (c0 + 32 < QKC);
    if (more) stage_next(smB[cur ^ 1], kbb + (size_t)(c0 + 32) * N_SP + m0, tid, wv);
    const v16bf a0 = load_a(qTb + (size_t)n0 * QKC + c0, QKC, lane);
    const v16bf a1 = load_a(qTb + (size_t)(n0 + 16) * QKC + c0, QKC, lane);
#pragma unroll
    for (int s = 0; s < 4; ++s) {
      const v16bf bf = lds_b(smB[cur], lane, s);
      acc[0][s] = wmma_bf16(a0, bf, acc[0][s]);
      acc[1][s] = wmma_bf16(a1, bf, acc[1][s]);
    }
    stage_sync(more, wv);
  }
  const int hi = (lane >> 4) & 1;
  const int lo = lane & 15;
#pragma unroll
  for (int t = 0; t < 2; ++t) {
#pragma unroll
    for (int s = 0; s < 4; ++s) {
#pragma unroll
      for (int r = 0; r < 8; ++r) {
        const int n = n0 + t * 16 + r + hi * 8;   // C-tile M index = n
        const int m = m0 + s * 16 + lo;           // C-tile N index = m
        float v = fmaxf(acc[t][s][r], 0.0f);
        Atb[(size_t)m * N_SP + n] = (__bf16)(v * v);   // transposed store
      }
    }
  }
}

// ---------------------------------------------------------------------------
// Kernel 3 (hot loop, ~87 GFLOP): Vg[h][n] = gate * sum_m v[h][m] * At[m][n].
// 256-thread block: 8 waves = 8 consecutive 32-row h-tiles, all sharing the
// TDM-staged 32x64 At B-tile (double-buffered; DMA overlapped with the
// 8 WMMA/wave of the current step). 72 K-steps.
// ---------------------------------------------------------------------------
__global__ void __launch_bounds__(256)
attnv_kernel(const __bf16* __restrict__ vb, const __bf16* __restrict__ At,
             const __bf16* __restrict__ gb, __bf16* __restrict__ Vg) {
  const int tid  = threadIdx.x;
  const int lane = tid & 31;
  const int wv   = tid >> 5;                        // 0..7
  const int n0 = blockIdx.x * 64;                   // gridDim.x = 36
  const int h0 = blockIdx.y * 256 + wv * 32;        // gridDim.y = 2
  const int b  = blockIdx.z;
  const __bf16* vbb = vb + (size_t)b * HIDD * N_SP;
  const __bf16* Atb = At + (size_t)b * N_SP * N_SP;
  __shared__ __align__(32) __bf16 smB[2][32 * 64];

  v8f acc[2][4] = {};
  stage_first(smB[0], Atb + n0, tid, wv);
  for (int m0 = 0; m0 < N_SP; m0 += 32) {
    const int cur = (m0 >> 5) & 1;
    const bool more = (m0 + 32 < N_SP);
    if (more)
      stage_next(smB[cur ^ 1], Atb + (size_t)(m0 + 32) * N_SP + n0, tid, wv);
    const v16bf a0 = load_a(vbb + (size_t)h0 * N_SP + m0, N_SP, lane);
    const v16bf a1 = load_a(vbb + (size_t)(h0 + 16) * N_SP + m0, N_SP, lane);
#pragma unroll
    for (int s = 0; s < 4; ++s) {
      const v16bf bf = lds_b(smB[cur], lane, s);
      acc[0][s] = wmma_bf16(a0, bf, acc[0][s]);
      acc[1][s] = wmma_bf16(a1, bf, acc[1][s]);
    }
    stage_sync(more, wv);
  }
  const int hi = (lane >> 4) & 1;
  const int lo = lane & 15;
#pragma unroll
  for (int t = 0; t < 2; ++t) {
#pragma unroll
    for (int s = 0; s < 4; ++s) {
#pragma unroll
      for (int r = 0; r < 8; ++r) {
        const int h = h0 + t * 16 + r + hi * 8;
        const int n = n0 + s * 16 + lo;
        const size_t idx = ((size_t)b * HIDD + h) * N_SP + n;
        const float g = (float)gb[idx];
        Vg[idx] = (__bf16)(acc[t][s][r] * g);
      }
    }
  }
}

// ---------------------------------------------------------------------------
// Kernel 4: out = x + b_out + w_out @ Vg.
// 256-thread block: 8 waves x 32 o-rows = the whole O=256, so each Vg B-tile
// is DMA'd from L2 exactly once per (n-block, batch). K = 512 -> 16 steps.
// ---------------------------------------------------------------------------
__global__ void __launch_bounds__(256)
out_kernel(const float* __restrict__ x, const float* __restrict__ w_out,
           const float* __restrict__ b_out, const __bf16* __restrict__ Vg,
           float* __restrict__ out) {
  const int tid  = threadIdx.x;
  const int lane = tid & 31;
  const int wv   = tid >> 5;                        // 0..7
  const int n0 = blockIdx.x * 64;                   // gridDim.x = 36
  const int o0 = wv * 32;                           // covers O = 256
  const int b  = blockIdx.z;
  const __bf16* Vgb = Vg + (size_t)b * HIDD * N_SP;
  __shared__ __align__(32) __bf16 smB[2][32 * 64];

  v8f acc[2][4] = {};
  stage_first(smB[0], Vgb + n0, tid, wv);
#pragma unroll 2
  for (int k0 = 0; k0 < HIDD; k0 += 32) {
    const int cur = (k0 >> 5) & 1;
    const bool more = (k0 + 32 < HIDD);
    if (more)
      stage_next(smB[cur ^ 1], Vgb + (size_t)(k0 + 32) * N_SP + n0, tid, wv);
    const v16bf a0 = load_a(w_out + (size_t)o0 * HIDD + k0, HIDD, lane);
    const v16bf a1 = load_a(w_out + (size_t)(o0 + 16) * HIDD + k0, HIDD, lane);
#pragma unroll
    for (int s = 0; s < 4; ++s) {
      const v16bf bf = lds_b(smB[cur], lane, s);
      acc[0][s] = wmma_bf16(a0, bf, acc[0][s]);
      acc[1][s] = wmma_bf16(a1, bf, acc[1][s]);
    }
    stage_sync(more, wv);
  }
  const int hi = (lane >> 4) & 1;
  const int lo = lane & 15;
#pragma unroll
  for (int t = 0; t < 2; ++t) {
#pragma unroll
    for (int s = 0; s < 4; ++s) {
#pragma unroll
      for (int r = 0; r < 8; ++r) {
        const int o = o0 + t * 16 + r + hi * 8;
        const int n = n0 + s * 16 + lo;
        const size_t idx = ((size_t)b * C_IN + o) * N_SP + n;
        out[idx] = x[idx] + b_out[o] + acc[t][s][r];
      }
    }
  }
}

// ---------------------------------------------------------------------------
// Workspace layout (bf16, 256B aligned slabs), total ~298 MB:
//   qT [B][N][QK], k [B][QK][N], v [B][HID][N], gate [B][HID][N],
//   Vg [B][HID][N], At [B][N][N] (170 MB; the 10.6 MB per-batch slice is
//   L2-resident while its batch is being consumed).
// ---------------------------------------------------------------------------
extern "C" void kernel_launch(void* const* d_in, const int* in_sizes, int n_in,
                              void* d_out, int out_size, void* d_ws, size_t ws_size,
                              hipStream_t stream) {
  const float* x        = (const float*)d_in[0];
  const float* w_hidden = (const float*)d_in[1];
  const float* b_hidden = (const float*)d_in[2];
  const float* w_qk     = (const float*)d_in[3];
  const float* b_qk     = (const float*)d_in[4];
  const float* w_out    = (const float*)d_in[5];
  const float* b_out    = (const float*)d_in[6];
  float* out = (float*)d_out;
  (void)in_sizes; (void)n_in; (void)out_size; (void)ws_size;

  char* base = (char*)d_ws;
  size_t off = 0;
  auto take = [&](size_t elems) {
    __bf16* p = (__bf16*)(base + off);
    off += (elems * sizeof(__bf16) + 255) & ~(size_t)255;
    return p;
  };
  __bf16* qT = take((size_t)B_SZ * N_SP * QKC);
  __bf16* kb = take((size_t)B_SZ * QKC * N_SP);
  __bf16* vb = take((size_t)B_SZ * HIDD * N_SP);
  __bf16* gb = take((size_t)B_SZ * HIDD * N_SP);
  __bf16* Vg = take((size_t)B_SZ * HIDD * N_SP);
  __bf16* At = take((size_t)B_SZ * N_SP * N_SP);

  proj_kernel <<<dim3(36, 76, B_SZ), dim3(32),  0, stream>>>(x, w_hidden, b_hidden,
                                                             w_qk, b_qk, qT, kb, vb, gb);
  sim_kernel  <<<dim3(9, 36, B_SZ),  dim3(256), 0, stream>>>(qT, kb, At);
  attnv_kernel<<<dim3(36, 2, B_SZ),  dim3(256), 0, stream>>>(vb, At, gb, Vg);
  out_kernel  <<<dim3(36, 1, B_SZ),  dim3(256), 0, stream>>>(x, w_out, b_out, Vg, out);
}
